// FrequencyAttention_88613765251420
// MI455X (gfx1250) — matrix operand order repaired and checked by reading, not compiled, not run on hardware
//
#include <hip/hip_runtime.h>
#include <math.h>

// FrequencyAttention for MI455X (gfx1250, wave32, WMMA).
// B=8, L=4096, D=1024, H=16, d_head=64, F=2049, k_top=64.

#define B_   8
#define L_   4096
#define D_   1024
#define H_   16
#define DH_  64
#define F_   2049
#define MTOT (B_*L_)     // 32768
#define TWO_PI 6.2831853071795864769f
#define LPAD 40          // LDS row stride (elems): 80B = 16B-multiple, conflict-light

typedef __attribute__((ext_vector_type(16))) __bf16 v16bf;
typedef __attribute__((ext_vector_type(8)))  float  v8f;

union Frag { unsigned int u[8]; v16bf v; };

// ---------------------------------------------------------------------------
// GEMM: C[M,N] = A[M,K] @ W[K,N] + bias[N], fp32 in/out, bf16 hi/lo split
// (3 WMMAs per tile: ah*bh + ah*bl + al*bh) for ~fp32 accuracy on matrix cores.
// Block tile 128x128, k-tile 32 (== bf16 WMMA K). 8 waves: 4(M) x 2(N),
// each wave owns a 32x64 C sub-tile = 2x4 WMMA fragments.
// Double-buffered LDS + register staging pipeline + L2 prefetch.
// ---------------------------------------------------------------------------
__global__ __launch_bounds__(256)
void gemm_bias_bf16x3(const float* __restrict__ A, const float* __restrict__ W,
                      const float* __restrict__ bias, float* __restrict__ C,
                      int M, int N, int K) {
  __shared__ __align__(16) __bf16 Ah[2][128][LPAD];
  __shared__ __align__(16) __bf16 Al[2][128][LPAD];
  __shared__ __align__(16) __bf16 Bh[2][128][LPAD];   // stored transposed: [n][k]
  __shared__ __align__(16) __bf16 Bl[2][128][LPAD];

  const int tid  = threadIdx.x;
  const int lane = tid & 31, wave = tid >> 5;
  const int wm = wave >> 1, wn = wave & 1;          // wave grid 4x2
  const int r = lane & 15, hlf = lane >> 4;         // lane row / half
  const int m0 = blockIdx.y * 128, n0 = blockIdx.x * 128;

  v8f acc[2][4];
  for (int i = 0; i < 2; i++)
    for (int j = 0; j < 4; j++)
      for (int e = 0; e < 8; e++) acc[i][j][e] = 0.0f;

  float aReg[16], wReg[16];
  // Prologue: stage k-tile 0 into registers.
  #pragma unroll
  for (int i = 0; i < 16; i++) {
    int e = tid + 256 * i;
    aReg[i] = A[(size_t)(m0 + (e >> 5)) * K + (e & 31)];
    wReg[i] = W[(size_t)(e >> 7) * N + n0 + (e & 127)];
  }

  int buf = 0;
  for (int kt = 0; kt < K; kt += 32) {
    // Commit staged tile to LDS (hi/lo bf16 split).
    #pragma unroll
    for (int i = 0; i < 16; i++) {
      int e = tid + 256 * i;
      {
        int row = e >> 5, col = e & 31;
        float v = aReg[i];
        __bf16 h = (__bf16)v;
        Ah[buf][row][col] = h;
        Al[buf][row][col] = (__bf16)(v - (float)h);
      }
      {
        int kk = e >> 7, nn = e & 127;
        float v = wReg[i];
        __bf16 h = (__bf16)v;
        Bh[buf][nn][kk] = h;
        Bl[buf][nn][kk] = (__bf16)(v - (float)h);
      }
    }
    __syncthreads();

    // Stage next k-tile while this one computes; prefetch tile after next to L2.
    if (kt + 32 < K) {
      #pragma unroll
      for (int i = 0; i < 16; i++) {
        int e = tid + 256 * i;
        aReg[i] = A[(size_t)(m0 + (e >> 5)) * K + (kt + 32) + (e & 31)];
        wReg[i] = W[(size_t)(kt + 32 + (e >> 7)) * N + n0 + (e & 127)];
      }
      if (kt + 64 < K) {
        __builtin_prefetch(&A[(size_t)(m0 + (tid >> 1)) * K + kt + 64], 0, 1);
        __builtin_prefetch(&W[(size_t)(kt + 64 + (tid >> 3)) * N + n0 + ((tid & 7) << 4)], 0, 1);
      }
    }

    // Build fragments per ISA VGPR layouts (wave32).
    Frag fah[2], fal[2], fbh[4], fbl[4];
    #pragma unroll
    for (int mf = 0; mf < 2; mf++) {
      const unsigned int* ph = (const unsigned int*)&Ah[buf][32 * wm + 16 * mf + r][0];
      const unsigned int* pl = (const unsigned int*)&Al[buf][32 * wm + 16 * mf + r][0];
      #pragma unroll
      for (int v2 = 0; v2 < 8; v2++) {
        int dwi = v2 + (v2 >= 4 ? 4 : 0) + 4 * hlf;   // A: interleaved k-pairs
        fah[mf].u[v2] = ph[dwi];
        fal[mf].u[v2] = pl[dwi];
      }
    }
    #pragma unroll
    for (int nf = 0; nf < 4; nf++) {
      const unsigned int* ph = (const unsigned int*)&Bh[buf][64 * wn + 16 * nf + r][0];
      const unsigned int* pl = (const unsigned int*)&Bl[buf][64 * wn + 16 * nf + r][0];
      #pragma unroll
      for (int v2 = 0; v2 < 8; v2++) {
        int dwi = v2 + (v2 >= 4 ? 4 : 0) + 4 * hlf;   // B: interleaved k-pairs
        fbh[nf].u[v2] = ph[dwi];
        fbl[nf].u[v2] = pl[dwi];
      }
    }

    #pragma unroll
    for (int mf = 0; mf < 2; mf++)
      #pragma unroll
      for (int nf = 0; nf < 4; nf++) {
        acc[mf][nf] = __builtin_amdgcn_wmma_f32_16x16x32_bf16(
            false, fah[mf].v, false, fbh[nf].v, (short)0, acc[mf][nf], false, false);
        acc[mf][nf] = __builtin_amdgcn_wmma_f32_16x16x32_bf16(
            false, fah[mf].v, false, fbl[nf].v, (short)0, acc[mf][nf], false, false);
        acc[mf][nf] = __builtin_amdgcn_wmma_f32_16x16x32_bf16(
            false, fal[mf].v, false, fbh[nf].v, (short)0, acc[mf][nf], false, false);
      }
    buf ^= 1;
  }

  // Epilogue: C fragment layout: VGPR i -> M = i + 8*half, N = lane&15
  #pragma unroll
  for (int mf = 0; mf < 2; mf++)
    #pragma unroll
    for (int nf = 0; nf < 4; nf++) {
      int n = n0 + 64 * wn + 16 * nf + r;
      float bn = bias[n];
      #pragma unroll
      for (int i = 0; i < 8; i++) {
        int m = m0 + 32 * wm + 16 * mf + i + 8 * hlf;
        C[(size_t)m * N + n] = acc[mf][nf][i] + bn;
      }
    }
}

// ---------------------------------------------------------------------------
// Transpose (B,L,D) -> (B,D,L) so FFT columns are contiguous.
// ---------------------------------------------------------------------------
__global__ __launch_bounds__(256)
void transpose_bld_bdl(const float* __restrict__ in, float* __restrict__ out) {
  __shared__ float t[32][33];
  int lb = blockIdx.x * 32, db = blockIdx.y * 32, b = blockIdx.z;
  int x = threadIdx.x & 31, y = threadIdx.x >> 5;   // 8 rows per pass
  #pragma unroll
  for (int i = 0; i < 4; i++) {
    int l = lb + y + 8 * i;
    t[y + 8 * i][x] = in[((size_t)b * L_ + l) * D_ + db + x];
  }
  __syncthreads();
  #pragma unroll
  for (int i = 0; i < 4; i++) {
    int d = db + y + 8 * i;
    out[((size_t)b * D_ + d) * L_ + lb + x] = t[x][y + 8 * i];
  }
}

// ---------------------------------------------------------------------------
// In-place radix-2 DIT complex FFT (len 4096, real input, imag=0) in LDS.
// One workgroup per (b, d) column. Emits Re/Im spectrum bins 0..2048 (== rfft)
// plus the per-column spectral mean (complex).
// ---------------------------------------------------------------------------
__global__ __launch_bounds__(256)
void fft_q(const float* __restrict__ Qt,
           float* __restrict__ ReQ, float* __restrict__ ImQ,
           float* __restrict__ qm_re, float* __restrict__ qm_im) {
  __shared__ float2 buf[4096];
  __shared__ float2 red[256];
  const int col = blockIdx.x;              // b*1024 + d
  const int tid = threadIdx.x;
  const float* src = Qt + (size_t)col * L_;

  for (int i = tid; i < 4096; i += 256) {
    int rj = (int)(__brev((unsigned)i) >> 20);     // 12-bit bit reversal
    buf[i] = make_float2(src[rj], 0.0f);
  }
  __syncthreads();

  for (int len = 2; len <= 4096; len <<= 1) {
    int half = len >> 1;
    float ang0 = -TWO_PI / (float)len;             // forward: e^{-i...}
    for (int t = tid; t < 2048; t += 256) {
      int g = t / half;
      int j = t - g * half;
      int i0 = g * len + j;
      int i1 = i0 + half;
      float a = ang0 * (float)j;
      float wc = cosf(a), wsn = sinf(a);
      float2 u = buf[i0], v = buf[i1];
      float vr = v.x * wc - v.y * wsn;
      float vi = v.x * wsn + v.y * wc;
      buf[i0] = make_float2(u.x + vr, u.y + vi);
      buf[i1] = make_float2(u.x - vr, u.y - vi);
    }
    __syncthreads();
  }

  float2 a2 = make_float2(0.0f, 0.0f);
  for (int f = tid; f < F_; f += 256) {
    float2 X = buf[f];
    ReQ[(size_t)col * F_ + f] = X.x;
    ImQ[(size_t)col * F_ + f] = X.y;
    a2.x += X.x; a2.y += X.y;
  }
  red[tid] = a2;
  __syncthreads();
  for (int off = 128; off > 0; off >>= 1) {
    if (tid < off) { red[tid].x += red[tid + off].x; red[tid].y += red[tid + off].y; }
    __syncthreads();
  }
  if (tid == 0) {
    qm_re[col] = red[0].x / (float)F_;
    qm_im[col] = red[0].y / (float)F_;
  }
}

// ---------------------------------------------------------------------------
// energy[b,h,f] = sum_dh |Q_freq| (deterministic; no atomics).
// ---------------------------------------------------------------------------
__global__ __launch_bounds__(256)
void energy_reduce(const float* __restrict__ ReQ, const float* __restrict__ ImQ,
                   float* __restrict__ energy) {
  int f = blockIdx.x * 256 + threadIdx.x;
  int bh = blockIdx.y;                    // b*16 + h
  if (f >= F_) return;
  int b = bh >> 4, h = bh & 15;
  float s = 0.0f;
  for (int dh = 0; dh < 64; dh++) {
    size_t base = ((size_t)(b * D_ + h * 64 + dh)) * F_ + f;
    float re = ReQ[base], im = ImQ[base];
    s += sqrtf(re * re + im * im);
  }
  energy[(size_t)bh * F_ + f] = s;
}

// ---------------------------------------------------------------------------
// Top-64 bins per (b,h) by iterative parallel argmax (ties -> lower index).
// Order of selected bins is irrelevant downstream (softmax + per-bin scatter).
// ---------------------------------------------------------------------------
__global__ __launch_bounds__(256)
void topk64(const float* __restrict__ energy, int* __restrict__ idx) {
  __shared__ float e[F_];
  __shared__ float rv[256];
  __shared__ int   ri[256];
  int bh = blockIdx.x, tid = threadIdx.x;
  for (int f = tid; f < F_; f += 256) e[f] = energy[(size_t)bh * F_ + f];
  __syncthreads();
  for (int k = 0; k < 64; k++) {
    float bv = -__builtin_inff(); int bi = 0;
    for (int f = tid; f < F_; f += 256) {
      float v = e[f];
      if (v > bv) { bv = v; bi = f; }
    }
    rv[tid] = bv; ri[tid] = bi;
    __syncthreads();
    for (int off = 128; off > 0; off >>= 1) {
      if (tid < off) {
        if (rv[tid + off] > rv[tid] ||
            (rv[tid + off] == rv[tid] && ri[tid + off] < ri[tid])) {
          rv[tid] = rv[tid + off]; ri[tid] = ri[tid + off];
        }
      }
      __syncthreads();
    }
    if (tid == 0) { idx[bh * 64 + k] = ri[0]; e[ri[0]] = -__builtin_inff(); }
    __syncthreads();
  }
}

// ---------------------------------------------------------------------------
// Per (b,h): Re K_freq / Re V_freq at selected bins via direct cosine DFT
// (exact angle reduction via (f*l)&4095), scores, softmax, src = attn * Vr.
// ---------------------------------------------------------------------------
__global__ __launch_bounds__(256)
void attn_kernel(const float* __restrict__ Kp, const float* __restrict__ Vp,
                 const float* __restrict__ ReQ, const int* __restrict__ idx,
                 float* __restrict__ srcout) {
  __shared__ float costab[4096];
  __shared__ float Kr[64][64];
  __shared__ float Vr[64][64];
  __shared__ int   fj[64];
  __shared__ float sc[64];
  int bh = blockIdx.x, tid = threadIdx.x;
  int b = bh >> 4, h = bh & 15;
  for (int i = tid; i < 4096; i += 256)
    costab[i] = cosf(TWO_PI * (float)i / 4096.0f);
  if (tid < 64) fj[tid] = idx[bh * 64 + tid];
  __syncthreads();

  const int dh = tid & 63;
  const int j0 = tid >> 6;                 // 0..3; thread covers j = j0 + 4*i
  float ak[16], av[16];
  int fr[16];
  #pragma unroll
  for (int i = 0; i < 16; i++) { ak[i] = 0.0f; av[i] = 0.0f; fr[i] = fj[j0 + 4 * i]; }

  const float* kcol = Kp + (size_t)b * L_ * D_ + h * 64 + dh;
  const float* vcol = Vp + (size_t)b * L_ * D_ + h * 64 + dh;
  for (int l = 0; l < 4096; l++) {
    float kv = kcol[(size_t)l * D_];
    float vv = vcol[(size_t)l * D_];
    #pragma unroll
    for (int i = 0; i < 16; i++) {
      float c = costab[(fr[i] * l) & 4095];
      ak[i] += kv * c; av[i] += vv * c;
    }
  }
  #pragma unroll
  for (int i = 0; i < 16; i++) { Kr[j0 + 4 * i][dh] = ak[i]; Vr[j0 + 4 * i][dh] = av[i]; }
  __syncthreads();

  if (tid < 64) {
    int j = tid, f = fj[j];
    float s = 0.0f;
    for (int d2 = 0; d2 < 64; d2++) {
      float qr = ReQ[((size_t)(b * D_ + h * 64 + d2)) * F_ + f];
      s += qr * Kr[j][d2];
    }
    sc[j] = s / (8.0f + 1e-8f);            // sqrt(d_head) + eps
  }
  __syncthreads();

  float mx = -__builtin_inff();
  for (int j = 0; j < 64; j++) mx = fmaxf(mx, sc[j]);
  float den = 0.0f;
  for (int j = 0; j < 64; j++) den += expf(sc[j] - mx);
  for (int p = tid; p < 4096; p += 256) {
    int j = p >> 6, d2 = p & 63;
    float a = expf(sc[j] - mx) / den;
    srcout[((size_t)bh * 64 + j) * 64 + d2] = a * Vr[j][d2];
  }
}

// ---------------------------------------------------------------------------
// Analytic irfft of (mean-filled spectrum with 64 scattered real bins):
//   x[l,dh] = cbase(l,dh) + (2/L)*Im(c)*sum_j sin(th_j) + sum_j A[j,dh]*cos(th_j)
//   A[j,dh] = w_j * (src[j,dh] - Re(c));  w_j = 1/L at f in {0, L/2} else 2/L
//   cbase: l==0 -> Re(c); odd l -> -(2/L)*Im(c)*cot(pi*l/L); else 0
// (C2R irfft discards Im at DC/Nyquist; sin vanishes there automatically.)
// ---------------------------------------------------------------------------
__global__ __launch_bounds__(256)
void inverse_kernel(const float* __restrict__ srcin, const int* __restrict__ idx,
                    const float* __restrict__ qm_re, const float* __restrict__ qm_im,
                    float* __restrict__ OT /* (B,L,D) */) {
  __shared__ float Amat[64][64];
  __shared__ float Bv[64], Rc[64];
  __shared__ int fj[64];
  int bh = blockIdx.x, tid = threadIdx.x;
  int b = bh >> 4, h = bh & 15;
  if (tid < 64) {
    fj[tid] = idx[bh * 64 + tid];
    Rc[tid] = qm_re[b * D_ + h * 64 + tid];
    Bv[tid] = (2.0f / 4096.0f) * qm_im[b * D_ + h * 64 + tid];
  }
  __syncthreads();
  for (int p = tid; p < 4096; p += 256) {
    int j = p >> 6, d2 = p & 63;
    int f = fj[j];
    float w = (f == 0 || f == 2048) ? (1.0f / 4096.0f) : (2.0f / 4096.0f);
    Amat[j][d2] = w * (srcin[((size_t)bh * 64 + j) * 64 + d2] - Rc[d2]);
  }
  __syncthreads();

  for (int l = tid; l < 4096; l += 256) {
    float acc2[64];
    #pragma unroll
    for (int d2 = 0; d2 < 64; d2++) acc2[d2] = 0.0f;
    float S = 0.0f;
    for (int j = 0; j < 64; j++) {
      int m = (fj[j] * l) & 4095;
      float th = TWO_PI * (float)m / 4096.0f;
      float cs = cosf(th), sn = sinf(th);
      S += sn;
      #pragma unroll
      for (int d2 = 0; d2 < 64; d2++) acc2[d2] += cs * Amat[j][d2];
    }
    float cot = 0.0f;
    if (l & 1) {
      float a = 3.14159265358979323846f * (float)l / 4096.0f;
      cot = cosf(a) / sinf(a);
    }
    float* orow = OT + ((size_t)b * L_ + l) * D_ + h * 64;
    #pragma unroll
    for (int d2 = 0; d2 < 64; d2++) {
      float cb = (l == 0) ? Rc[d2] : ((l & 1) ? (-Bv[d2] * cot) : 0.0f);
      orow[d2] = cb + Bv[d2] * S + acc2[d2];
    }
  }
}

// ---------------------------------------------------------------------------
extern "C" void kernel_launch(void* const* d_in, const int* in_sizes, int n_in,
                              void* d_out, int out_size, void* d_ws, size_t ws_size,
                              hipStream_t stream) {
  (void)in_sizes; (void)n_in; (void)out_size; (void)ws_size;
  const float* query = (const float*)d_in[0];
  const float* Wq = (const float*)d_in[1];
  const float* bq = (const float*)d_in[2];
  const float* Wk = (const float*)d_in[3];
  const float* bk = (const float*)d_in[4];
  const float* Wv = (const float*)d_in[5];
  const float* bv = (const float*)d_in[6];
  const float* Wo = (const float*)d_in[7];
  const float* bo = (const float*)d_in[8];
  float* out = (float*)d_out;

  float* ws = (float*)d_ws;
  size_t off = 0;
  float* Q    = ws + off; off += (size_t)MTOT * D_;        // 128 MB
  float* Kp   = ws + off; off += (size_t)MTOT * D_;        // 128 MB
  float* Vp   = ws + off; off += (size_t)MTOT * D_;        // 128 MB
  float* Qt   = ws + off; off += (size_t)MTOT * D_;        // 128 MB (reused as out_time)
  float* ReQ  = ws + off; off += (size_t)B_ * D_ * F_;     // 67 MB
  float* ImQ  = ws + off; off += (size_t)B_ * D_ * F_;     // 67 MB
  float* qmr  = ws + off; off += (size_t)B_ * D_;
  float* qmi  = ws + off; off += (size_t)B_ * D_;
  float* eng  = ws + off; off += (size_t)B_ * H_ * F_;
  float* srcb = ws + off; off += (size_t)B_ * H_ * 64 * 64;
  int*   idx  = (int*)(ws + off); off += (size_t)B_ * H_ * 64;

  dim3 gg(D_ / 128, MTOT / 128);   // (8, 256)

  gemm_bias_bf16x3<<<gg, 256, 0, stream>>>(query, Wq, bq, Q,  MTOT, D_, D_);
  gemm_bias_bf16x3<<<gg, 256, 0, stream>>>(query, Wk, bk, Kp, MTOT, D_, D_);
  gemm_bias_bf16x3<<<gg, 256, 0, stream>>>(query, Wv, bv, Vp, MTOT, D_, D_);
  transpose_bld_bdl<<<dim3(L_ / 32, D_ / 32, B_), 256, 0, stream>>>(Q, Qt);
  fft_q<<<B_ * D_, 256, 0, stream>>>(Qt, ReQ, ImQ, qmr, qmi);
  energy_reduce<<<dim3((F_ + 255) / 256, B_ * H_), 256, 0, stream>>>(ReQ, ImQ, eng);
  topk64<<<B_ * H_, 256, 0, stream>>>(eng, idx);
  attn_kernel<<<B_ * H_, 256, 0, stream>>>(Kp, Vp, ReQ, idx, srcb);
  float* OT = Qt;  // Qt dead after fft_q; reuse as out_time (B,L,D)
  inverse_kernel<<<B_ * H_, 256, 0, stream>>>(srcb, idx, qmr, qmi, OT);
  gemm_bias_bf16x3<<<gg, 256, 0, stream>>>(OT, Wo, bo, out, MTOT, D_, D_);
}
